// Attention4D_41669772706636
// MI455X (gfx1250) — matrix operand order repaired and checked by reading, not compiled, not run on hardware
//
#include <hip/hip_runtime.h>
#include <hip/hip_bf16.h>

typedef __attribute__((ext_vector_type(16))) _Float16 v16h;
typedef __attribute__((ext_vector_type(8)))  _Float16 h8;
typedef __attribute__((ext_vector_type(8)))  float    v8f;

#define NHEADS 8
#define KDIM   32
#define DVDIM  128
#define CDIM   384
#define NTOK   196     // 14*14
#define NPAD   208     // 13*16
#define MPAD   224     // 7*32  (K-padding for probs@v)
#define BATCH  64
#define CH     1024    // NHEADS*DVDIM
#define ATTN_SCALE 0.17677669529663687f

static __device__ __forceinline__ v8f wmma_f16(v16h a, v16h b, v8f c) {
  // 8 args: (neg_a, A, neg_b, B, c_mod, C, reuse_a, reuse_b)
  return __builtin_amdgcn_wmma_f32_16x16x32_f16(false, a, false, b, (short)0, c,
                                                false, false);
}
static __device__ __forceinline__ v16h cat16(h8 lo, h8 hi) {
  return __builtin_shufflevector(lo, hi, 0,1,2,3,4,5,6,7,8,9,10,11,12,13,14,15);
}

// ---------------------------------------------------------------------------
// Prep kernels: f32 -> f16 layout shuffles (pure bandwidth, ~30 MB total)
// ---------------------------------------------------------------------------
__global__ void xt_kernel(const float* __restrict__ x, _Float16* __restrict__ xt)
{
  const int idx = blockIdx.x * blockDim.x + threadIdx.x;   // c fastest
  if (idx >= BATCH * NTOK * CDIM) return;
  const int c = idx % CDIM;
  const int n = (idx / CDIM) % NTOK;
  const int b = idx / (CDIM * NTOK);
  xt[idx] = (_Float16)x[((size_t)b * CDIM + c) * NTOK + n];
}

__global__ void wcat_kernel(const float* __restrict__ qw, const float* __restrict__ kw,
                            const float* __restrict__ vw, _Float16* __restrict__ wcat)
{
  const int idx = blockIdx.x * blockDim.x + threadIdx.x;
  if (idx >= 1536 * CDIM) return;
  const int row = idx / CDIM, col = idx % CDIM;
  float v;
  if (row < 256)      v = qw[(size_t)row * CDIM + col];
  else if (row < 512) v = kw[(size_t)(row - 256) * CDIM + col];
  else                v = vw[(size_t)(row - 512) * CDIM + col];
  wcat[idx] = (_Float16)v;
}

__global__ void wproj_kernel(const float* __restrict__ projw, _Float16* __restrict__ wp)
{
  const int idx = blockIdx.x * blockDim.x + threadIdx.x;
  if (idx >= CDIM * CH) return;
  wp[idx] = (_Float16)projw[idx];
}

// ---------------------------------------------------------------------------
// Kernel 1: fused QKV projection.  M=1536, N=196, K=384.
// One wave per 16x16 tile, 12 K-steps; all fragments are contiguous loads.
// ---------------------------------------------------------------------------
__global__ void qkv_proj_kernel(const _Float16* __restrict__ wcat,
                                const _Float16* __restrict__ xt,
                                const float* __restrict__ qb, const float* __restrict__ kb,
                                const float* __restrict__ vb,
                                _Float16* __restrict__ qf, _Float16* __restrict__ kt,
                                _Float16* __restrict__ vt, float* __restrict__ v4)
{
  const int mt   = blockIdx.x;         // 0..95
  const int nt   = blockIdx.y;         // 0..12
  const int b    = blockIdx.z;
  const int lane = threadIdx.x;        // wave32
  const int g    = lane >> 4;
  const int lm   = lane & 15;
  const int m0   = mt * 16;
  const int n0   = nt * 16;

  const _Float16* wrow = wcat + (size_t)(m0 + lm) * CDIM;   // A row = out channel
  int ncol = n0 + lm; if (ncol > NTOK - 1) ncol = NTOK - 1;
  const _Float16* xrow = xt + ((size_t)b * NTOK + ncol) * CDIM;  // B col = token

  v8f acc = {};
  for (int k0 = 0; k0 < CDIM; k0 += 32) {
    v16h a  = *(const v16h*)(wrow + k0 + g * 16);
    v16h bf = *(const v16h*)(xrow + k0 + g * 16);
    acc = wmma_f16(a, bf, acc);
  }

  const int nc = n0 + lm;            // D column (token)
  if (nc < NTOK) {
#pragma unroll
    for (int r = 0; r < 8; ++r) {
      const int o = m0 + r + 8 * g;  // D row (channel)
      float val = acc[r];
      if (o < 256) {                                   // q -> [bh][n][kd]
        val += qb[o];
        const int h = o >> 5, kd = o & 31;
        qf[((size_t)(b * NHEADS + h) * NPAD + nc) * KDIM + kd] = (_Float16)val;
      } else if (o < 512) {                            // k -> [bh][m][kd]  (row-major!)
        const int oo = o - 256;
        val += kb[oo];
        const int h = oo >> 5, kd = oo & 31;
        kt[((size_t)(b * NHEADS + h) * NPAD + nc) * KDIM + kd] = (_Float16)val;
      } else {                                         // v -> f32 [b][n][ch] + f16 [bh][dv][m]
        const int ch = o - 512;
        val += vb[ch];
        v4[((size_t)b * NTOK + nc) * CH + ch] = val;
        const int h = ch >> 7, dv = ch & 127;
        vt[((size_t)(b * NHEADS + h) * DVDIM + dv) * MPAD + nc] = (_Float16)val;
      }
    }
  }
}

// ---------------------------------------------------------------------------
// Kernel 2: fused attention.  grid = (13 n-tiles, 64 batch), block = 256
// (8 waves = 8 heads).  bias + th1 + softmax + th2 all stay in LDS.
// ---------------------------------------------------------------------------
__global__ void attn_kernel(const _Float16* __restrict__ qf,
                            const _Float16* __restrict__ kt,
                            const _Float16* __restrict__ vt,
                            const float* __restrict__ th1_w, const float* __restrict__ th1_b,
                            const float* __restrict__ th2_w, const float* __restrict__ th2_b,
                            const float* __restrict__ bias_seg,
                            const int*   __restrict__ bias_idxs,
                            float* __restrict__ oat)
{
  extern __shared__ char smem[];
  float*    lds  = (float*)smem;                         // [8*16][MPAD] f32 logits
  _Float16* pf16 = (_Float16*)(smem + 8 * 16 * MPAD * 4); // [8*16][MPAD] f16 probs
  __shared__ float s_w1[64], s_w2[64], s_b1[8], s_b2[8];

  const int nt   = blockIdx.x;
  const int b    = blockIdx.y;
  const int tid  = threadIdx.x;
  const int h    = tid >> 5;                     // wave id == head
  const int lane = tid & 31;
  const int g    = lane >> 4;
  const int lm   = lane & 15;
  const int n0   = nt * 16;

  if (tid < 64) { s_w1[tid] = th1_w[tid]; s_w2[tid] = th2_w[tid]; }
  if (tid < 8)  { s_b1[tid] = th1_b[tid]; s_b2[tid] = th2_b[tid]; }

  // ---- stage 1: logits = scale * q@k^T + bias -----------------------------
  const _Float16* qbase = qf + ((size_t)(b * NHEADS + h) * NPAD + (n0 + lm)) * KDIM;
  v16h a = *(const v16h*)(qbase + g * 16);

  const _Float16* kbase = kt + (size_t)(b * NHEADS + h) * NPAD * KDIM;
  float* lrow = lds + (size_t)h * 16 * MPAD;

  for (int mt = 0; mt < 13; ++mt) {
    const int m0 = mt * 16;
    // B fragment: column m = m0+lm, K = kd contiguous
    v16h bf = *(const v16h*)(kbase + (size_t)(m0 + lm) * KDIM + g * 16);
    v8f c = {};
    c = wmma_f16(a, bf, c);

    const int m = m0 + lm;
#pragma unroll
    for (int r = 0; r < 8; ++r) {
      const int row = r + 8 * g;
      const int n   = n0 + row;
      float v = 0.f;
      if (m < NTOK) {
        v = c[r] * ATTN_SCALE;
        if (n < NTOK) v += bias_seg[h * NTOK + bias_idxs[n * NTOK + m]];
      }
      lrow[row * MPAD + m] = v;
    }
  }
  __syncthreads();

  // ---- stage 2a: talking-heads mix #1 (cross-head, in place) --------------
  for (int pos = tid; pos < 16 * MPAD; pos += 256) {
    const int r = pos / MPAD, m = pos % MPAD;
    if (m < NTOK) {
      float vals[NHEADS];
#pragma unroll
      for (int gg = 0; gg < NHEADS; ++gg) vals[gg] = lds[(gg * 16 + r) * MPAD + m];
#pragma unroll
      for (int hh = 0; hh < NHEADS; ++hh) {
        float s = s_b1[hh];
#pragma unroll
        for (int gg = 0; gg < NHEADS; ++gg) s += s_w1[hh * 8 + gg] * vals[gg];
        lds[(hh * 16 + r) * MPAD + m] = s;
      }
    }
  }
  __syncthreads();

  // ---- stage 2b: softmax over valid columns (one thread per (h,row)) ------
  if (tid < 128) {
    float* base = lds + tid * MPAD;
    float mx = -1e30f;
    for (int m = 0; m < NTOK; ++m) mx = fmaxf(mx, base[m]);
    float s = 0.f;
    for (int m = 0; m < NTOK; ++m) { float e = __expf(base[m] - mx); base[m] = e; s += e; }
    const float inv = 1.f / s;
    for (int m = 0; m < NTOK; ++m) base[m] *= inv;
  }
  __syncthreads();

  // ---- stage 2c: talking-heads mix #2 -> f16 probs; zero padded columns ---
  for (int pos = tid; pos < 16 * MPAD; pos += 256) {
    const int r = pos / MPAD, m = pos % MPAD;
    if (m < NTOK) {
      float vals[NHEADS];
#pragma unroll
      for (int gg = 0; gg < NHEADS; ++gg) vals[gg] = lds[(gg * 16 + r) * MPAD + m];
#pragma unroll
      for (int hh = 0; hh < NHEADS; ++hh) {
        float s = s_b2[hh];
#pragma unroll
        for (int gg = 0; gg < NHEADS; ++gg) s += s_w2[hh * 8 + gg] * vals[gg];
        pf16[(hh * 16 + r) * MPAD + m] = (_Float16)s;
      }
    } else {
#pragma unroll
      for (int hh = 0; hh < NHEADS; ++hh) pf16[(hh * 16 + r) * MPAD + m] = (_Float16)0.f;
    }
  }
  __syncthreads();

  // ---- stage 3: o = probs @ v  (16x128 per head, K padded to 224) ---------
  const _Float16* vbase = vt + (size_t)(b * NHEADS + h) * DVDIM * MPAD;
  const _Float16* prow  = pf16 + (size_t)h * 16 * MPAD;
  for (int dt = 0; dt < 8; ++dt) {
    const int dv0 = dt * 16;
    v8f acc = {};
    for (int k0 = 0; k0 < MPAD; k0 += 32) {
      v16h pa = *(const v16h*)(prow + lm * MPAD + k0 + g * 16);
      v16h vb = *(const v16h*)(vbase + (size_t)(dv0 + lm) * MPAD + k0 + g * 16);
      acc = wmma_f16(pa, vb, acc);
    }
#pragma unroll
    for (int r = 0; r < 8; ++r) {
      const int n = n0 + r + 8 * g;
      if (n < NTOK)
        oat[((size_t)b * NTOK + n) * CH + h * DVDIM + dv0 + lm] = acc[r];
    }
  }
}

// ---------------------------------------------------------------------------
// Kernel 3: depthwise 3x3 conv + attn output + ReLU -> f16 [b][n][ch]
// ---------------------------------------------------------------------------
__global__ void conv_relu_kernel(const float* __restrict__ v4,
                                 const float* __restrict__ vlw,
                                 const float* __restrict__ vlb,
                                 const float* __restrict__ oat,
                                 _Float16* __restrict__ oact)
{
  const int idx = blockIdx.x * blockDim.x + threadIdx.x;   // ch fastest
  if (idx >= BATCH * NTOK * CH) return;
  const int ch = idx & (CH - 1);
  const int n  = (idx >> 10) % NTOK;
  const int bb = idx / (CH * NTOK);
  const int y = n / 14, xq = n % 14;
  const float* src = v4 + (size_t)bb * NTOK * CH + ch;
  float s = vlb[ch];
#pragma unroll
  for (int ky = 0; ky < 3; ++ky) {
    const int yy = y + ky - 1;
    if (yy < 0 || yy >= 14) continue;
#pragma unroll
    for (int kx = 0; kx < 3; ++kx) {
      const int xx = xq + kx - 1;
      if (xx < 0 || xx >= 14) continue;
      s += src[(size_t)(yy * 14 + xx) * CH] * vlw[ch * 9 + ky * 3 + kx];
    }
  }
  float o = oat[idx] + s;
  o = fmaxf(o, 0.f);
  oact[idx] = (_Float16)o;
}

// ---------------------------------------------------------------------------
// Kernel 4: output projection (384x1024)@(1024x196) per batch
// ---------------------------------------------------------------------------
__global__ void out_proj_kernel(const _Float16* __restrict__ oact,
                                const _Float16* __restrict__ wproj,
                                const float* __restrict__ projb,
                                float* __restrict__ out)
{
  const int mt   = blockIdx.x;       // 0..23
  const int nt   = blockIdx.y;       // 0..12
  const int b    = blockIdx.z;
  const int lane = threadIdx.x;
  const int g    = lane >> 4;
  const int lm   = lane & 15;
  const int m0   = mt * 16, n0 = nt * 16;

  const _Float16* wrow = wproj + (size_t)(m0 + lm) * CH;
  int ncol = n0 + lm; if (ncol > NTOK - 1) ncol = NTOK - 1;
  const _Float16* ob = oact + ((size_t)b * NTOK + ncol) * CH;

  v8f acc = {};
  for (int k0 = 0; k0 < CH; k0 += 32) {
    v16h a  = *(const v16h*)(wrow + k0 + g * 16);
    v16h bf = *(const v16h*)(ob + k0 + g * 16);
    acc = wmma_f16(a, bf, acc);
  }

  const int nc = n0 + lm;
  if (nc < NTOK) {
#pragma unroll
    for (int r = 0; r < 8; ++r) {
      const int o = m0 + r + 8 * g;
      out[((size_t)b * CDIM + o) * NTOK + nc] = acc[r] + projb[o];
    }
  }
}

// ---------------------------------------------------------------------------
extern "C" void kernel_launch(void* const* d_in, const int* in_sizes, int n_in,
                              void* d_out, int out_size, void* d_ws, size_t ws_size,
                              hipStream_t stream)
{
  (void)in_sizes; (void)n_in; (void)out_size; (void)ws_size;
  const float* x        = (const float*)d_in[0];
  const float* qw       = (const float*)d_in[1];
  const float* qb       = (const float*)d_in[2];
  const float* kw       = (const float*)d_in[3];
  const float* kb       = (const float*)d_in[4];
  const float* vw       = (const float*)d_in[5];
  const float* vb       = (const float*)d_in[6];
  const float* vlw      = (const float*)d_in[7];
  const float* vlb      = (const float*)d_in[8];
  const float* th1w     = (const float*)d_in[9];
  const float* th1b     = (const float*)d_in[10];
  const float* th2w     = (const float*)d_in[11];
  const float* th2b     = (const float*)d_in[12];
  const float* projw    = (const float*)d_in[13];
  const float* projb    = (const float*)d_in[14];
  const float* bias_seg = (const float*)d_in[15];
  const int*   bias_idx = (const int*)d_in[16];

  // f16 scratch regions
  const size_t XTE = (size_t)BATCH * NTOK * CDIM;           // x transposed
  const size_t WCE = (size_t)1536 * CDIM;                   // qkv weights
  const size_t WPE = (size_t)CDIM * CH;                     // proj weights
  const size_t QE  = (size_t)BATCH * NHEADS * NPAD * KDIM;  // q
  const size_t KE  = (size_t)BATCH * NHEADS * NPAD * KDIM;  // k (row-major)
  const size_t VE  = (size_t)BATCH * NHEADS * DVDIM * MPAD; // v transposed
  const size_t OE  = (size_t)BATCH * NTOK * CH;             // f32 each (v4, oat)

  _Float16* xt    = (_Float16*)d_ws;
  _Float16* wcat  = xt + XTE;
  _Float16* wproj = wcat + WCE;
  _Float16* qf    = wproj + WPE;
  _Float16* kt    = qf + QE;
  _Float16* vt    = kt + KE;
  float*    v4    = (float*)(vt + VE);
  float*    oat   = v4 + OE;
  _Float16* oact  = (_Float16*)(oat + OE);

  xt_kernel<<<(int)((XTE + 255) / 256), 256, 0, stream>>>(x, xt);
  wcat_kernel<<<(int)((WCE + 255) / 256), 256, 0, stream>>>(qw, kw, vw, wcat);
  wproj_kernel<<<(int)((WPE + 255) / 256), 256, 0, stream>>>(projw, wproj);

  qkv_proj_kernel<<<dim3(96, 13, BATCH), 32, 0, stream>>>(
      wcat, xt, qb, kb, vb, qf, kt, vt, v4);

  const int attn_lds = 8 * 16 * MPAD * (int)(sizeof(float) + sizeof(_Float16));
  attn_kernel<<<dim3(13, BATCH), 256, attn_lds, stream>>>(
      qf, kt, vt, th1w, th1b, th2w, th2b, bias_seg, bias_idx, oat);

  conv_relu_kernel<<<(int)((OE + 255) / 256), 256, 0, stream>>>(
      v4, vlw, vlb, oat, oact);

  out_proj_kernel<<<dim3(24, 13, BATCH), 32, 0, stream>>>(
      oact, wproj, projb, (float*)d_out);
}
